// GraphNet_38079180046870
// MI455X (gfx1250) — matrix operand order repaired and checked by reading, not compiled, not run on hardware
//
#include <hip/hip_runtime.h>

// ---------------------------------------------------------------------------
// MI455X (gfx1250) GraphNet implementation.
// GEMMs via v_wmma_f32_16x16x32_bf16, wave32, 4 waves/block, 64 rows/block.
// Weights pre-swizzled (RNE) into a contiguous-k (k = 16h+s) B-fragment layout; activations kept
// in LDS as bf16 (truncated) in the A-fragment-friendly row-major layout.
// A-fragments are hoisted into registers once per layer (reused across all
// 8 N-tiles) so LDS traffic is dominated by B-fragment b128 loads.
// ---------------------------------------------------------------------------

typedef __bf16 v16bf __attribute__((ext_vector_type(16)));
typedef float  v8f   __attribute__((ext_vector_type(8)));

__device__ __forceinline__ unsigned f2bf_rne(float f) {   // prep path only
  unsigned u = __float_as_uint(f);
  return (u + 0x7FFFu + ((u >> 16) & 1u)) >> 16;
}
__device__ __forceinline__ unsigned short f2bf_trunc(float f) {  // hot path
  return (unsigned short)(__float_as_uint(f) >> 16);
}
__device__ __forceinline__ float bf2f(unsigned short b) {
  return __uint_as_float(((unsigned)b) << 16);
}

struct Gather4 {
  const float* p0; const int* i0; int w0;
  const float* p1; const int* i1; int w1;
  const float* p2; const int* i2; int w2;
  const float* p3; const int* i3; int w3;
};

// ------------------------- weight pre-swizzle ------------------------------
// Packs W[K][N] (fp32, row-major) into bf16 pairs laid out as
// dst[((kt*Np + n)*2 + h)*8 + j] = { W[kt*32+16h+2j][n], W[..+1][n] }
// i.e. per-lane k = 16h+s (contiguous order), matched by the A loads in gemm_stage.
__global__ void k_prep(const float* __restrict__ W, int K, int N, int KT, int Np,
                       unsigned* __restrict__ dst) {
  long t = (long)blockIdx.x * blockDim.x + threadIdx.x;
  long total = (long)KT * Np * 16;
  if (t >= total) return;
  int j  = (int)(t & 7);
  int h  = (int)((t >> 3) & 1);
  int n  = (int)((t >> 4) % Np);
  int kt = (int)(t / (16L * Np));
  int k0 = kt * 32 + 16 * h + 2 * j;
  float f0 = (k0     < K && n < N) ? W[(long)k0 * N + n]       : 0.f;
  float f1 = (k0 + 1 < K && n < N) ? W[(long)(k0 + 1) * N + n] : 0.f;
  dst[t] = f2bf_rne(f0) | (f2bf_rne(f1) << 16);
}

// ------------------------------ zero fill ----------------------------------
__global__ void k_zero(float* __restrict__ p, long n) {
  long i = (long)blockIdx.x * blockDim.x + threadIdx.x;
  if (i < n) p[i] = 0.f;
}

// ------------------------- one WMMA GEMM layer -----------------------------
// act[64 x (KT*32)] (bf16 LDS) @ W[(KT*32) x (NT*16)] (swizzled LDS)
// -> bf16 LDS out, with bias (+ optional leaky relu). One 16-row M-tile/wave.
template <int KT, int NT, bool RELU>
__device__ __forceinline__ void gemm_stage(
    const unsigned* __restrict__ actIn, int pitchHalf,     // uints per row
    const unsigned* __restrict__ W, const float* __restrict__ bias,
    unsigned short* __restrict__ o16, int pitchOut,
    int lane, int m0) {
  const int h   = lane >> 4;
  const int n15 = lane & 15;
  union AB { unsigned u[8]; v16bf v; };

  // Hoist all A fragments for this wave's M-tile (reused across N-tiles).
  AB afr[KT];
  const unsigned* ar = actIn + (m0 + n15) * pitchHalf;
#pragma unroll
  for (int kt = 0; kt < KT; ++kt) {
#pragma unroll
    for (int j = 0; j < 8; ++j) {  // A layout matching B: K = 16h+2j
      int k0 = 16 * h + 2 * j;
      afr[kt].u[j] = ar[kt * 16 + (k0 >> 1)];
    }
  }
  constexpr int Np = NT * 16;
  for (int nt = 0; nt < NT; ++nt) {
    v8f acc = {0.f, 0.f, 0.f, 0.f, 0.f, 0.f, 0.f, 0.f};
#pragma unroll
    for (int kt = 0; kt < KT; ++kt) {
      AB b;
      const unsigned* br = W + (((kt * Np) + (nt * 16 + n15)) * 2 + h) * 8;
#pragma unroll
      for (int j = 0; j < 8; ++j) b.u[j] = br[j];
      acc = __builtin_amdgcn_wmma_f32_16x16x32_bf16(
          false, afr[kt].v, false, b.v, (short)0, acc, false, false);
    }
    int n = nt * 16 + n15;
    float bb = bias[n];
#pragma unroll
    for (int r = 0; r < 8; ++r) {    // D layout: lane=col, VGPR r -> M=r+8h
      float v = acc[r] + bb;
      if (RELU) v = (v > 0.f) ? v : 0.01f * v;
      o16[(m0 + r + 8 * h) * pitchOut + n] = f2bf_trunc(v);
    }
  }
}

// --------------------------- fused WMMA MLP --------------------------------
// y = MLP(x): L1(K1->128, lrelu), L2(128->128, lrelu), L3(128->NOUT)
// optional LayerNorm(g,beta), optional residual, optional atomic scatter-sum.
template <int K1P, int NOUTP, bool LN, bool RES, bool SCAT>
__global__ __launch_bounds__(128) void k_mlp(
    Gather4 gs, int K1, int NOUT, int nRows,
    const unsigned* __restrict__ w1g, const unsigned* __restrict__ w2g,
    const unsigned* __restrict__ w3g,
    const float* __restrict__ b1, const float* __restrict__ b2,
    const float* __restrict__ b3,
    const float* __restrict__ lng, const float* __restrict__ lnb,
    const float* __restrict__ res, float* __restrict__ out, int outPitch,
    float* __restrict__ scat, const int* __restrict__ scatIdx) {
  constexpr int KT1 = K1P / 32;
  constexpr int AP  = (K1P > 128) ? K1P : 128;
  static_assert(K1P % 32 == 0 && NOUTP % 16 == 0, "pad shapes");

  __shared__ unsigned sW1[KT1 * 2048];      // swizzled bf16 pairs
  __shared__ unsigned sW2[4 * 2048];
  __shared__ unsigned sW3[4 * NOUTP * 16];
  __shared__ unsigned sA[64 * AP / 2];      // bf16 activations (buffer A)
  __shared__ unsigned sB[64 * 128 / 2];     // bf16 activations (buffer B)
  __shared__ float sB1[128], sB2[128], sB3[NOUTP], sG[NOUTP], sBt[NOUTP];

  const int tid = threadIdx.x;
  for (int t = tid; t < KT1 * 2048; t += 128) sW1[t] = w1g[t];
  for (int t = tid; t < 4 * 2048; t += 128)   sW2[t] = w2g[t];
  for (int t = tid; t < 4 * NOUTP * 16; t += 128) sW3[t] = w3g[t];
  { sB1[tid] = b1[tid]; sB2[tid] = b2[tid]; }
  if (tid < NOUTP) {
    sB3[tid] = (tid < NOUT) ? b3[tid] : 0.f;
    sG[tid]  = (LN && tid < NOUT) ? lng[tid] : 1.f;
    sBt[tid] = (LN && tid < NOUT) ? lnb[tid] : 0.f;
  }

  // Build input tile (gathered concat), stored bf16, zero-padded.
  const int rowBase = blockIdx.x * 64;
  unsigned short* a16 = (unsigned short*)sA;
  for (int t = tid; t < 64 * K1P; t += 128) {
    int r = t / K1P, c = t - r * K1P;
    int gr = rowBase + r;
    float v = 0.f;
    if (gr < nRows && c < K1) {
      int cc = c;
      if (cc < gs.w0) {
        int rr = gs.i0 ? gs.i0[gr] : gr; v = gs.p0[(long)rr * gs.w0 + cc];
      } else if ((cc -= gs.w0) < gs.w1) {
        int rr = gs.i1 ? gs.i1[gr] : gr; v = gs.p1[(long)rr * gs.w1 + cc];
      } else if ((cc -= gs.w1) < gs.w2) {
        int rr = gs.i2 ? gs.i2[gr] : gr; v = gs.p2[(long)rr * gs.w2 + cc];
      } else if ((cc -= gs.w2) < gs.w3) {
        int rr = gs.i3 ? gs.i3[gr] : gr; v = gs.p3[(long)rr * gs.w3 + cc];
      }
    }
    a16[r * K1P + c] = f2bf_trunc(v);
  }
  __syncthreads();

  const int lane = tid & 31;          // wave32
  const int m0   = (tid >> 5) * 16;   // 16-row M-tile per wave

  gemm_stage<KT1, 8, true >(sA, K1P / 2, sW1, sB1, (unsigned short*)sB, 128, lane, m0);
  __syncthreads();
  gemm_stage<4, 8, true >(sB, 64, sW2, sB2, (unsigned short*)sA, 128, lane, m0);
  __syncthreads();
  gemm_stage<4, NOUTP / 16, false>(sA, 64, sW3, sB3, (unsigned short*)sB, NOUTP, lane, m0);
  __syncthreads();

  // LayerNorm + residual + store (+ optional segment-sum scatter).
  if (tid < 64) {
    int gr = rowBase + tid;
    if (gr < nRows) {
      const unsigned short* y = (const unsigned short*)sB + tid * NOUTP;
      float mu = 0.f, inv = 1.f;
      if (LN) {
        for (int c = 0; c < NOUT; ++c) mu += bf2f(y[c]);
        mu /= (float)NOUT;
        float var = 0.f;
        for (int c = 0; c < NOUT; ++c) { float d = bf2f(y[c]) - mu; var += d * d; }
        var /= (float)NOUT;
        inv = rsqrtf(var + 1e-5f);
      }
      for (int c = 0; c < NOUT; ++c) {
        float v = bf2f(y[c]);
        if (LN) v = (v - mu) * inv * sG[c] + sBt[c];
        if (RES) v += res[(long)gr * NOUT + c];
        out[(long)gr * outPitch + c] = v;
        if (SCAT) atomicAdd(&scat[(long)scatIdx[gr] * NOUT + c], v);
      }
    }
  }
}

// ------------------- tiny boundary encoder (7->16->16->8) ------------------
__global__ __launch_bounds__(256) void k_enc_small(
    const float* __restrict__ ef, const float* __restrict__ nf,
    const int* __restrict__ sidx,
    const float* __restrict__ W1, const float* __restrict__ b1,
    const float* __restrict__ W2, const float* __restrict__ b2,
    const float* __restrict__ W3, const float* __restrict__ b3,
    const float* __restrict__ g, const float* __restrict__ bt,
    float* __restrict__ out, int n) {
  int i = blockIdx.x * blockDim.x + threadIdx.x;
  if (i >= n) return;
  float x[7];
#pragma unroll
  for (int k = 0; k < 4; ++k) x[k] = ef[i * 4 + k];
  int s = sidx[i];
#pragma unroll
  for (int k = 0; k < 3; ++k) x[4 + k] = nf[s * 3 + k];
  float h1[16];
#pragma unroll
  for (int o = 0; o < 16; ++o) {
    float a = b1[o];
#pragma unroll
    for (int k = 0; k < 7; ++k) a += x[k] * W1[k * 16 + o];
    h1[o] = (a > 0.f) ? a : 0.01f * a;
  }
  float h2[16];
  for (int o = 0; o < 16; ++o) {
    float a = b2[o];
    for (int k = 0; k < 16; ++k) a += h1[k] * W2[k * 16 + o];
    h2[o] = (a > 0.f) ? a : 0.01f * a;
  }
  float y[8]; float mu = 0.f;
  for (int o = 0; o < 8; ++o) {
    float a = b3[o];
    for (int k = 0; k < 16; ++k) a += h2[k] * W3[k * 8 + o];
    y[o] = a; mu += a;
  }
  mu *= 0.125f;
  float var = 0.f;
  for (int o = 0; o < 8; ++o) { float d = y[o] - mu; var += d * d; }
  var *= 0.125f;
  float inv = rsqrtf(var + 1e-5f);
  for (int o = 0; o < 8; ++o) out[i * 8 + o] = (y[o] - mu) * inv * g[o] + bt[o];
}

// ------------------------ macro flow correction ----------------------------
__global__ void k_macro_sum(const float* __restrict__ h, const int* __restrict__ s_i2m,
                            const int* __restrict__ d_i2m, float* mS, float* mC, int n) {
  int e = blockIdx.x * blockDim.x + threadIdx.x;
  if (e >= n) return;
  float q = h[(long)s_i2m[e] * 2 + 1];
  atomicAdd(&mS[d_i2m[e]], q);
  atomicAdd(&mC[d_i2m[e]], 1.f);
}
__global__ void k_corr(const float* __restrict__ h, const int* __restrict__ s_m2i,
                       const int* __restrict__ d_m2i, const float* __restrict__ mS,
                       const float* __restrict__ mC, float* cS, float* cC, int n) {
  int e = blockIdx.x * blockDim.x + threadIdx.x;
  if (e >= n) return;
  int m = s_m2i[e];
  float avg = mS[m] / fmaxf(mC[m], 1.f);
  float ce = avg - h[(long)d_m2i[e] * 2 + 1];
  atomicAdd(&cS[d_m2i[e]], ce);
  atomicAdd(&cC[d_m2i[e]], 1.f);
}
__global__ void k_final(const float* __restrict__ h, const float* __restrict__ cS,
                        const float* __restrict__ cC, float* __restrict__ out, int n) {
  int i = blockIdx.x * blockDim.x + threadIdx.x;
  if (i >= n) return;
  out[i * 2]     = h[i * 2];
  out[i * 2 + 1] = h[i * 2 + 1] + cS[i] / fmaxf(cC[i], 1.f);
}

// ------------------------------- host --------------------------------------
// d_in order: top-level setup_inputs() insertion order, with `params`
// expanded at position 6 as jax tree_flatten leaves (sorted dict keys):
//  0..5 : in_feat, e_inner_feat, e_in_feat, n_in_feat, e_out_feat, n_out_feat
//  6..11: dec      (hidW,hidb,inW,inb,outW,outb)
// 12..19: enc_edges(beta,g,hidW,hidb,inW,inb,outW,outb)
// 20..27: enc_in   (beta,g,hidW,hidb,inW,inb,outW,outb)
// 28..35: enc_nodes(beta,g,hidW,hidb,inW,inb,outW,outb)
// 36..43: enc_out  (beta,g,hidW,hidb,inW,inb,outW,outb)
// 44..67: proc_e[0..2] (8 each)   68..91: proc_n[0..2] (8 each)
// 92..101: src_inner,dst_inner,src_in,dst_in,src_out,dst_out,
//          src_i2m,dst_i2m,src_m2i,dst_m2i
extern "C" void kernel_launch(void* const* d_in, const int* in_sizes, int n_in,
                              void* d_out, int out_size, void* d_ws, size_t ws_size,
                              hipStream_t stream) {
  (void)n_in; (void)out_size; (void)ws_size;
  auto Fp = [&](int i) { return (const float*)d_in[i]; };
  auto Ip = [&](int i) { return (const int*)d_in[i]; };

  const int N  = in_sizes[0] / 12;   // 300000
  const int E  = in_sizes[1] / 4;    // 600000
  const int NM = N / 100;

  char* wp = (char*)d_ws;
  auto alloc = [&](size_t bytes) -> void* {
    void* p = (void*)wp;
    wp += (bytes + 255) & ~(size_t)255;
    return p;
  };
  float* procN  = (float*)alloc((size_t)N * 64 * 4);
  float* procE  = (float*)alloc((size_t)E * 64 * 4);
  float* peSum  = (float*)alloc((size_t)N * 64 * 4);
  float* inlet  = (float*)alloc((size_t)N * 8 * 4);
  float* outlet = (float*)alloc((size_t)N * 8 * 4);
  float* hbuf   = (float*)alloc((size_t)N * 2 * 4);
  float* mS = (float*)alloc((size_t)NM * 4);
  float* mC = (float*)alloc((size_t)NM * 4);
  float* cS = (float*)alloc((size_t)N * 4);
  float* cC = (float*)alloc((size_t)N * 4);

  auto allocW = [&](int Kp, int Np) {
    return (unsigned*)alloc((size_t)(Kp / 32) * Np * 16 * 4);
  };
  unsigned *encN_w1 = allocW(32, 128),  *encN_w2 = allocW(128, 128), *encN_w3 = allocW(128, 64);
  unsigned *encE_w1 = allocW(32, 128),  *encE_w2 = allocW(128, 128), *encE_w3 = allocW(128, 64);
  unsigned *pe_w1[3], *pe_w2[3], *pe_w3[3], *pn_w1[3], *pn_w2[3], *pn_w3[3];
  for (int j = 0; j < 3; ++j) {
    pe_w1[j] = allocW(192, 128); pe_w2[j] = allocW(128, 128); pe_w3[j] = allocW(128, 64);
    pn_w1[j] = allocW(160, 128); pn_w2[j] = allocW(128, 128); pn_w3[j] = allocW(128, 64);
  }
  unsigned *dec_w1 = allocW(64, 128), *dec_w2 = allocW(128, 128), *dec_w3 = allocW(128, 16);

  auto prep = [&](const float* W, int K, int Nn, int Kp, int Np, unsigned* dst) {
    int KT = Kp / 32;
    long total = (long)KT * Np * 16;
    k_prep<<<(int)((total + 255) / 256), 256, 0, stream>>>(W, K, Nn, KT, Np, dst);
  };
  // encoders
  prep(Fp(32), 12, 128, 32, 128, encN_w1);  prep(Fp(30), 128, 128, 128, 128, encN_w2);
  prep(Fp(34), 128, 64, 128, 64, encN_w3);
  prep(Fp(16), 4, 128, 32, 128, encE_w1);   prep(Fp(14), 128, 128, 128, 128, encE_w2);
  prep(Fp(18), 128, 64, 128, 64, encE_w3);
  // processors
  for (int j = 0; j < 3; ++j) {
    int be = 44 + 8 * j, bn = 68 + 8 * j;
    prep(Fp(be + 4), 192, 128, 192, 128, pe_w1[j]);
    prep(Fp(be + 2), 128, 128, 128, 128, pe_w2[j]);
    prep(Fp(be + 6), 128, 64, 128, 64, pe_w3[j]);
    prep(Fp(bn + 4), 144, 128, 160, 128, pn_w1[j]);
    prep(Fp(bn + 2), 128, 128, 128, 128, pn_w2[j]);
    prep(Fp(bn + 6), 128, 64, 128, 64, pn_w3[j]);
  }
  // decoder
  prep(Fp(8), 64, 128, 64, 128, dec_w1);  prep(Fp(6), 128, 128, 128, 128, dec_w2);
  prep(Fp(10), 128, 2, 128, 16, dec_w3);

  const dim3 blk(128);
  const int gN = (N + 63) / 64, gE = (E + 63) / 64;

  // encode nodes / edges
  Gather4 g0{Fp(0), nullptr, 12, nullptr, nullptr, 0, nullptr, nullptr, 0, nullptr, nullptr, 0};
  k_mlp<32, 64, true, false, false><<<gN, blk, 0, stream>>>(
      g0, 12, 64, N, encN_w1, encN_w2, encN_w3, Fp(33), Fp(31), Fp(35), Fp(29), Fp(28),
      nullptr, procN, 64, nullptr, nullptr);
  Gather4 g1{Fp(1), nullptr, 4, nullptr, nullptr, 0, nullptr, nullptr, 0, nullptr, nullptr, 0};
  k_mlp<32, 64, true, false, false><<<gE, blk, 0, stream>>>(
      g1, 4, 64, E, encE_w1, encE_w2, encE_w3, Fp(17), Fp(15), Fp(19), Fp(13), Fp(12),
      nullptr, procE, 64, nullptr, nullptr);

  // boundary encoders (dst_in/dst_out are arange -> seg_sum is identity)
  k_enc_small<<<(N + 255) / 256, 256, 0, stream>>>(
      Fp(2), Fp(3), Ip(94), Fp(24), Fp(25), Fp(22), Fp(23), Fp(26), Fp(27),
      Fp(21), Fp(20), inlet, N);
  k_enc_small<<<(N + 255) / 256, 256, 0, stream>>>(
      Fp(4), Fp(5), Ip(96), Fp(40), Fp(41), Fp(38), Fp(39), Fp(42), Fp(43),
      Fp(37), Fp(36), outlet, N);

  // process iterations
  for (int it = 0; it < 3; ++it) {
    long nz = (long)N * 64;
    k_zero<<<(int)((nz + 255) / 256), 256, 0, stream>>>(peSum, nz);
    int be = 44 + 8 * it, bn = 68 + 8 * it;
    Gather4 ge{procE, nullptr, 64, procN, Ip(92), 64, procN, Ip(93), 64, nullptr, nullptr, 0};
    k_mlp<192, 64, true, true, true><<<gE, blk, 0, stream>>>(
        ge, 192, 64, E, pe_w1[it], pe_w2[it], pe_w3[it],
        Fp(be + 5), Fp(be + 3), Fp(be + 7), Fp(be + 1), Fp(be + 0),
        procE, procE, 64, peSum, Ip(93));
    Gather4 gn{procN, nullptr, 64, peSum, nullptr, 64, inlet, nullptr, 8, outlet, nullptr, 8};
    k_mlp<160, 64, true, true, false><<<gN, blk, 0, stream>>>(
        gn, 144, 64, N, pn_w1[it], pn_w2[it], pn_w3[it],
        Fp(bn + 5), Fp(bn + 3), Fp(bn + 7), Fp(bn + 1), Fp(bn + 0),
        procN, procN, 64, nullptr, nullptr);
  }

  // decode
  Gather4 gd{procN, nullptr, 64, nullptr, nullptr, 0, nullptr, nullptr, 0, nullptr, nullptr, 0};
  k_mlp<64, 16, false, false, false><<<gN, blk, 0, stream>>>(
      gd, 64, 2, N, dec_w1, dec_w2, dec_w3, Fp(9), Fp(7), Fp(11), nullptr, nullptr,
      nullptr, hbuf, 2, nullptr, nullptr);

  // macro-node average flowrate correction
  k_zero<<<(NM + 255) / 256, 256, 0, stream>>>(mS, NM);
  k_zero<<<(NM + 255) / 256, 256, 0, stream>>>(mC, NM);
  k_zero<<<(N + 255) / 256, 256, 0, stream>>>(cS, N);
  k_zero<<<(N + 255) / 256, 256, 0, stream>>>(cC, N);
  k_macro_sum<<<(N + 255) / 256, 256, 0, stream>>>(hbuf, Ip(98), Ip(99), mS, mC, N);
  k_corr<<<(N + 255) / 256, 256, 0, stream>>>(hbuf, Ip(100), Ip(101), mS, mC, cS, cC, N);
  k_final<<<(N + 255) / 256, 256, 0, stream>>>(hbuf, cS, cC, (float*)d_out, N);
}
